// CNNLR_58712202936614
// MI455X (gfx1250) — matrix operand order repaired and checked by reading, not compiled, not run on hardware
//
#include <hip/hip_runtime.h>

typedef __attribute__((ext_vector_type(16))) _Float16 v16h;
typedef __attribute__((ext_vector_type(8)))  _Float16 v8h;
typedef __attribute__((ext_vector_type(8)))  float    v8f;

#define B_    128
#define L_    26
#define C1_   128
#define C2_   64
#define K1_   7
#define K2_   5
#define NPOS  25
#define FEAT  (L_ * C2_)        // 1664
#define LP    30                // padded position slots: slot = l + 2, l in [-2..27]
#define KC2   (K2_ * C1_)       // 640 = conv2 GEMM K dimension (t*128 + c1)
#define NKC   (KC2 / 32)        // 20 K-chunks of 32
#define MROWS (B_ * L_)         // 3328 GEMM rows for conv2
#define NPAIR 325

// workspace layout (bytes); total ~4.93 MB
#define OFF_H1P  0                            // f16 [128][30][128]      983040
#define OFF_W2P  983040                       // f16  80 blocks * 1024    81920
#define OFF_WSP  (983040 + 81920)             // f16 2600 blocks * 1024 2662400
#define OFF_FH   (OFF_WSP + 2662400)          // f16 [128][26][64]       425984
#define OFF_FF   (OFF_FH + 425984)            // f32 [128][26][64]       851968
#define OFF_P2   (OFF_FF + 851968)            // f32 [100][128]           51200

// ---------------------------------------------------------------------------
// Kernel 0: one-shot f32->f16 weight packing into WMMA B-fragment order.
// B 32x16 layout: lane = N (c2/u), element e -> K = (lane/16)*8 + e for e<8,
// K = 16 + (lane/16)*8 + (e-8) for e>=8. Each lane's 16 halves stored contiguously
// (32 B), block = 1 KB, perfectly coalesced on both ends.
//   blocks [0, 80):        conv2 weights, block = kc*4 + ntile, K = t*128+c1
//   blocks [80, 80+2600):  bilinear W_{ij}, block = (pair*4 + ntile)*2 + kh
// ---------------------------------------------------------------------------
__global__ void __launch_bounds__(32) pack_kernel(
    const float* __restrict__ w2, const float* __restrict__ regw,
    _Float16* __restrict__ w2p, _Float16* __restrict__ wsp) {
  const int bid = blockIdx.x;
  const int lane = threadIdx.x, lg = lane & 15, hb = lane >> 4;
  if (bid < NKC * 4) {
    const int kc = bid >> 2, ntile = bid & 3;
    const int c2 = ntile * 16 + lg;
    _Float16* dst = w2p + bid * 512 + lane * 16;
#pragma unroll
    for (int e = 0; e < 16; ++e) {
      const int K = kc * 32 + hb * 8 + (e & 7) + (e >> 3) * 16;      // K = t*128 + c1
      const int t = K >> 7, c1 = K & 127;
      dst[e] = (_Float16)w2[(c2 * C1_ + c1) * K2_ + t];
    }
  } else {
    const int idx = bid - NKC * 4;              // pair*8 + ntile*2 + kh
    const int pairIdx = idx >> 3;
    const int ntile = (idx >> 1) & 3;
    const int kh = idx & 1;
    int p = pairIdx, i = 0;
    while (p >= NPOS - i) { p -= NPOS - i; ++i; }
    const int j = i + 1 + p;
    const long basei = 4096L * (25L * i - (long)i * (i - 1) / 2);
    const int rowlen = (NPOS - i) * C2_;
    const int u = ntile * 16 + lg;
    const float* src = regw + 1 + FEAT + basei +
                       (long)u * rowlen + (j - i - 1) * C2_ + kh * 32 + hb * 8;
    _Float16* dst = wsp + ((pairIdx * 4 + ntile) * 2 + kh) * 512 + lane * 16;
#pragma unroll
    for (int e = 0; e < 16; ++e) dst[e] = (_Float16)src[e + (e >> 3) * 8];
  }
}

// ---------------------------------------------------------------------------
// Kernel 1: conv1 over one-hot input = weight table lookup (VALU, trivial).
// Emits h1 transposed + position-padded as f16: h1p[n][l+2][c1], slots
// 0,1,28,29 zeroed so conv2's im2col A-loads need no boundary branches.
// ---------------------------------------------------------------------------
__global__ void __launch_bounds__(256) conv1_kernel(
    const int* __restrict__ x, const float* __restrict__ w1,
    const float* __restrict__ b1, _Float16* __restrict__ h1p) {
  __shared__ int sx[L_];
  const int n = blockIdx.x, tid = threadIdx.x;
  if (tid < L_) sx[tid] = x[n * L_ + tid];
  _Float16* row = h1p + n * (LP * C1_);
  for (int o = tid; o < 4 * C1_; o += 256) {       // zero pad slots 0,1,28,29
    const int s = o >> 7;
    const int slot = (s < 2) ? s : (26 + s);
    row[slot * C1_ + (o & 127)] = (_Float16)0.f;
  }
  __syncthreads();
  for (int o = tid; o < C1_ * L_; o += 256) {
    const int c = o / L_, l = o - c * L_;
    float acc = b1[c];
#pragma unroll
    for (int t = 0; t < K1_; ++t) {
      const int ll = l + t - K1_ / 2;
      if (ll >= 0 && ll < L_) acc += w1[c * 4 * K1_ + sx[ll] * K1_ + t];
    }
    row[(l + 2) * C1_ + c] = (_Float16)(acc > 0.f ? acc : 0.f);
  }
}

// ---------------------------------------------------------------------------
// Kernel 2: conv2 as im2col GEMM on WMMA: feat(3328 x 64) = A(3328 x 640) @ B.
// A[g=(n,l), K=t*128+c1] = h1p[n][l+t][c1] -> contiguous 16 B per 8-K run
// (runs are 8-aligned so they never straddle a tap t). One wave per
// (mtile, ntile); 20 v_wmma per wave. Epilogue adds bias, relu, writes feat
// in f32 and f16.
// ---------------------------------------------------------------------------
__global__ void __launch_bounds__(32) conv2_kernel(
    const _Float16* __restrict__ h1p, const _Float16* __restrict__ w2p,
    const float* __restrict__ b2,
    float* __restrict__ featF, _Float16* __restrict__ featH) {
  const int bid = blockIdx.x;                 // mtile*4 + ntile
  const int mtile = bid >> 2, ntile = bid & 3;
  const int lane = threadIdx.x, lg = lane & 15, hb = lane >> 4;
  const int g = mtile * 16 + lg;              // GEMM row = n*26 + l
  const int n = g / L_, l = g - n * L_;
  const _Float16* arow = h1p + n * (LP * C1_) + l * C1_;  // + t*128 + c1

  v8f acc = {};
  for (int kc = 0; kc < NKC; ++kc) {
    const int K0 = kc * 32 + hb * 8;          // first 8-K run of this lane
    const int K1 = K0 + 16;                   // second run
    const v8h a0 = *(const v8h*)(arow + (K0 >> 7) * C1_ + (K0 & 127));
    const v8h a1 = *(const v8h*)(arow + (K1 >> 7) * C1_ + (K1 & 127));
    v16h a;
#pragma unroll
    for (int e = 0; e < 8; ++e) { a[e] = a0[e]; a[8 + e] = a1[e]; }
    const v16h b = *(const v16h*)(w2p + (kc * 4 + ntile) * 512 + lane * 16);
    acc = __builtin_amdgcn_wmma_f32_16x16x32_f16(false, a, false, b,
                                                 (short)0, acc, false, false);
  }
  const int c2 = ntile * 16 + lg;
  const float bias = b2[c2];
#pragma unroll
  for (int r = 0; r < 8; ++r) {               // C layout: M = r + 8*half, N = lg
    const int gr = mtile * 16 + r + 8 * hb;
    const int nn = gr / L_, ll = gr - nn * L_;
    float v = acc[r] + bias;
    v = v > 0.f ? v : 0.f;
    featF[nn * FEAT + ll * C2_ + c2] = v;
    featH[nn * FEAT + ll * C2_ + c2] = (_Float16)v;
  }
}

// ---------------------------------------------------------------------------
// Kernel 3: second-order bilinear. One wave per (i, mtile, ntile):
// g_i tile = sum_{j>i} feat_j @ W_ij^T (pure b128 loads + WMMA now), then
// dot with feat_i, shfl_xor reduce over N, deterministic unique-slot store.
// ---------------------------------------------------------------------------
__global__ void __launch_bounds__(32) bilinear_kernel(
    const _Float16* __restrict__ featH, const float* __restrict__ featF,
    const _Float16* __restrict__ wsp, float* __restrict__ part2) {
  const int bid = blockIdx.x;
  const int i = bid >> 5;                     // 0..24
  const int mtile = (bid >> 2) & 7, ntile = bid & 3;
  const int lane = threadIdx.x, lg = lane & 15, hb = lane >> 4;
  const int pairBase = 25 * i - i * (i - 1) / 2;
  const int m = mtile * 16 + lg;
  const int u = ntile * 16 + lg;

  v8f acc = {};
  for (int j = i + 1; j <= NPOS; ++j) {
    const _Float16* wblk = wsp + (((pairBase + (j - i - 1)) * 4 + ntile) * 2) * 512;
    if (j < NPOS)                             // prefetch next pair's B blocks
      __builtin_prefetch(wblk + 8 * 512, 0, 1);
#pragma unroll
    for (int kh = 0; kh < 2; ++kh) {
      const _Float16* ap = featH + (m * L_ + j) * C2_ + kh * 32 + hb * 8;
      const v8h alo = *(const v8h*)ap;
      const v8h ahi = *(const v8h*)(ap + 16);
      v16h a;
#pragma unroll
      for (int e = 0; e < 8; ++e) { a[e] = alo[e]; a[8 + e] = ahi[e]; }
      const v16h b = *(const v16h*)(wblk + kh * 512 + lane * 16);
      acc = __builtin_amdgcn_wmma_f32_16x16x32_f16(false, a, false, b,
                                                   (short)0, acc, false, false);
    }
  }
  float total[8];
#pragma unroll
  for (int r = 0; r < 8; ++r) {
    const int n = mtile * 16 + r + 8 * hb;
    float p = acc[r] * featF[n * FEAT + i * C2_ + u];
    p += __shfl_xor(p, 1);
    p += __shfl_xor(p, 2);
    p += __shfl_xor(p, 4);
    p += __shfl_xor(p, 8);
    total[r] = p;
  }
  if (lg == 0) {
#pragma unroll
    for (int r = 0; r < 8; ++r) {
      const int n = mtile * 16 + r + 8 * hb;
      part2[(i * 4 + ntile) * B_ + n] = total[r];
    }
  }
}

// ---------------------------------------------------------------------------
// Kernel 4: out[n] = bias + w0 + <feat_n, w_first> + sum of 100 partials.
// ---------------------------------------------------------------------------
__global__ void __launch_bounds__(256) final_kernel(
    const float* __restrict__ featF, const float* __restrict__ regw,
    const float* __restrict__ regb, const float* __restrict__ part2,
    float* __restrict__ out) {
  __shared__ float red[256];
  const int n = blockIdx.x, tid = threadIdx.x;
  float s = 0.f;
  for (int idx = tid; idx < FEAT; idx += 256)
    s += featF[n * FEAT + idx] * regw[1 + idx];
  for (int idx = tid; idx < NPOS * 4; idx += 256)
    s += part2[idx * B_ + n];
  red[tid] = s;
  __syncthreads();
  for (int off = 128; off > 0; off >>= 1) {
    if (tid < off) red[tid] += red[tid + off];
    __syncthreads();
  }
  if (tid == 0) out[n] = red[0] + regw[0] + regb[0];
}

// ---------------------------------------------------------------------------
extern "C" void kernel_launch(void* const* d_in, const int* in_sizes, int n_in,
                              void* d_out, int out_size, void* d_ws, size_t ws_size,
                              hipStream_t stream) {
  const int*   x  = (const int*)d_in[0];
  const float* w1 = (const float*)d_in[1];
  const float* b1 = (const float*)d_in[2];
  const float* w2 = (const float*)d_in[3];
  const float* b2 = (const float*)d_in[4];
  const float* rw = (const float*)d_in[5];
  const float* rb = (const float*)d_in[6];
  float* out = (float*)d_out;

  char* ws = (char*)d_ws;
  _Float16* h1p   = (_Float16*)(ws + OFF_H1P);
  _Float16* w2p   = (_Float16*)(ws + OFF_W2P);
  _Float16* wsp   = (_Float16*)(ws + OFF_WSP);
  _Float16* featH = (_Float16*)(ws + OFF_FH);
  float*    featF = (float*)(ws + OFF_FF);
  float*    part2 = (float*)(ws + OFF_P2);

  pack_kernel<<<NKC * 4 + NPAIR * 8, 32, 0, stream>>>(w2, rw, w2p, wsp);
  conv1_kernel<<<B_, 256, 0, stream>>>(x, w1, b1, h1p);
  conv2_kernel<<<(MROWS / 16) * 4, 32, 0, stream>>>(h1p, w2p, b2, featF, featH);
  bilinear_kernel<<<NPOS * 32, 32, 0, stream>>>(featH, featF, wsp, part2);
  final_kernel<<<B_, 256, 0, stream>>>(featF, rw, rb, part2, out);
}
